// StabilizerEmbedding_21320217658053
// MI455X (gfx1250) — compile-verified, matrix operand order checked by reading, not run on hardware
//
#include <hip/hip_runtime.h>
#include <hip/hip_bf16.h>
#include <math.h>

// ---------------------------------------------------------------------------
// Fused StabilizerEmbedding + 2x (LN -> GEMM(256x256) -> GELU -> GEMM -> +res)
// Target: gfx1250 (CDNA5), wave32, v_wmma_f32_16x16x32_bf16
// MBLK=64 tokens/block: halves L2 weight traffic vs MBLK=32, 256 WMMAs/wave.
// ---------------------------------------------------------------------------

typedef __bf16 bf16;
typedef __attribute__((ext_vector_type(16))) __bf16 v16bf;
typedef __attribute__((ext_vector_type(8)))  float  v8f;

#define DHID   256
#define MBLK   64            // tokens per workgroup (4 M-tiles of 16)
#define XPITCH 260           // fp32 residual buffer pitch (pad vs bank conflicts)
#define APITCH 264           // bf16 operand buffer pitch (16B-aligned rows)
#define LN_EPS 1e-5f

union FragAB {
  v16bf v;
  uint4 q[2];
};

__device__ __forceinline__ void load_frag(FragAB& f, const bf16* p) {
  // two contiguous 16B chunks: K = base..base+7 and base+16..base+23
  f.q[0] = *(const uint4*)(p);
  f.q[1] = *(const uint4*)(p + 16);
}

// LayerNorm: xbuf(f32) -> abuf(bf16). One wave handles 8 tokens.
__device__ __forceinline__ void layernorm_ln(const float* xbufp, bf16* abufp,
                                             const float* __restrict__ g,
                                             const float* __restrict__ be) {
  const int wave = threadIdx.x >> 5;
  const int lane = threadIdx.x & 31;
#pragma unroll
  for (int i = 0; i < MBLK / 8; ++i) {
    const int t = wave * (MBLK / 8) + i;
    const float* row = xbufp + t * XPITCH;
    float s = 0.f, s2 = 0.f;
#pragma unroll
    for (int j = 0; j < 8; ++j) {
      const float v = row[lane + 32 * j];
      s += v; s2 += v * v;
    }
#pragma unroll
    for (int off = 16; off > 0; off >>= 1) {
      s  += __shfl_xor(s,  off, 32);
      s2 += __shfl_xor(s2, off, 32);
    }
    const float mu   = s * (1.0f / DHID);
    const float var  = s2 * (1.0f / DHID) - mu * mu;
    const float rinv = rsqrtf(var + LN_EPS);
#pragma unroll
    for (int j = 0; j < 8; ++j) {
      const int d = lane + 32 * j;
      abufp[t * APITCH + d] = (bf16)((row[d] - mu) * rinv * g[d] + be[d]);
    }
  }
}

// One 64x256 @ 256x256 GEMM over the block tile.
// GELU=true : out = gelu(acc + bias) -> abuf (bf16), after a block barrier
// GELU=false: xbuf += acc + bias (residual epilogue)
template <bool GELU>
__device__ __forceinline__ void gemm_tile(bf16* abufp, float* xbufp,
                                          const bf16* __restrict__ wt,
                                          const float* __restrict__ bias) {
  const int wave = threadIdx.x >> 5;
  const int lane = threadIdx.x & 31;
  const int hl   = lane >> 4;      // lane half (K-offset selector)
  const int m    = lane & 15;      // M index for A, N index for B/C/D

  v8f acc[2][4] = {};              // [n-tile][m-tile]

#pragma unroll
  for (int nti = 0; nti < 2; ++nti) {
    const int n = (wave * 2 + nti) * 16 + m;
    const bf16* wrow = wt + (size_t)n * DHID;   // W^T row: contiguous in K
#pragma unroll
    for (int k0 = 0; k0 < DHID; k0 += 32) {
      FragAB fb;
      load_frag(fb, wrow + k0 + hl * 8);        // global_load_b128 x2 (L2-hot)
      if (k0 + 32 < DHID)
        __builtin_prefetch(wrow + k0 + 32 + hl * 8, 0, 3);  // WGP-scope prefetch
#pragma unroll
      for (int mt = 0; mt < 4; ++mt) {
        FragAB fa;
        load_frag(fa, abufp + (mt * 16 + m) * APITCH + k0 + hl * 8); // ds_load_b128 x2
        acc[nti][mt] = __builtin_amdgcn_wmma_f32_16x16x32_bf16(
            false, fa.v, false, fb.v, (short)0, acc[nti][mt], false, false);
      }
    }
  }

  __syncthreads();   // all reads of abuf done before GELU path overwrites it

#pragma unroll
  for (int nti = 0; nti < 2; ++nti) {
    const int col = (wave * 2 + nti) * 16 + m;
    const float bv = bias[col];
#pragma unroll
    for (int mt = 0; mt < 4; ++mt) {
#pragma unroll
      for (int r = 0; r < 8; ++r) {
        const int row = mt * 16 + hl * 8 + r;   // C/D layout: M = r + 8*half, N = m
        float v = acc[nti][mt][r] + bv;
        if (GELU) {
          v = 0.5f * v * (1.0f + erff(v * 0.70710678118654752f)); // exact-erf GELU
          abufp[row * APITCH + col] = (bf16)v;
        } else {
          xbufp[row * XPITCH + col] += v;       // residual add
        }
      }
    }
  }
}

__global__ __launch_bounds__(256) void stab_embed_resmlp(
    const float* __restrict__ meas,  const float* __restrict__ evnt,
    const float* __restrict__ leak,  const float* __restrict__ evlk,
    const int*   __restrict__ sid,   const int*   __restrict__ cid,
    const float* __restrict__ w_meas, const float* __restrict__ b_meas,
    const float* __restrict__ w_event, const float* __restrict__ b_event,
    const float* __restrict__ w_leak,  const float* __restrict__ b_leak,
    const float* __restrict__ w_el,    const float* __restrict__ b_el,
    const float* __restrict__ stab_table, const float* __restrict__ cycle_table,
    const float* __restrict__ g1, const float* __restrict__ be1,
    const float* __restrict__ b11, const float* __restrict__ b12,
    const float* __restrict__ g2, const float* __restrict__ be2,
    const float* __restrict__ b21, const float* __restrict__ b22,
    const bf16* __restrict__ wt,    // [4][256][256] bf16, transposed (n-major)
    float* __restrict__ out) {
  __shared__ float xbuf[MBLK * XPITCH];   // 66,560 B
  __shared__ bf16  abuf[MBLK * APITCH];   // 33,792 B  (CDNA5: 320 KB LDS/WGP)

  const int tid  = threadIdx.x;           // == feature index d (blockDim=256)
  const int base = blockIdx.x * MBLK;

  // ---- fused embedding: 4 scalar projections + 2 table gathers ----
  {
    const float wm = w_meas[tid],  we = w_event[tid];
    const float wl = w_leak[tid],  w4 = w_el[tid];
    const float bsum = b_meas[tid] + b_event[tid] + b_leak[tid] + b_el[tid];
#pragma unroll 4
    for (int t = 0; t < MBLK; ++t) {
      const int gidx = base + t;
      float h = meas[gidx] * wm + evnt[gidx] * we +
                leak[gidx] * wl + evlk[gidx] * w4 + bsum;
      h += stab_table[(size_t)sid[gidx] * DHID + tid];
      h += cycle_table[(size_t)cid[gidx] * DHID + tid];
      xbuf[t * XPITCH + tid] = h;
    }
  }
  __syncthreads();

  // ---- res block 1 ----
  layernorm_ln(xbuf, abuf, g1, be1);
  __syncthreads();
  gemm_tile<true >(abuf, xbuf, wt + 0 * DHID * DHID, b11);
  __syncthreads();
  gemm_tile<false>(abuf, xbuf, wt + 1 * DHID * DHID, b12);
  __syncthreads();

  // ---- res block 2 ----
  layernorm_ln(xbuf, abuf, g2, be2);
  __syncthreads();
  gemm_tile<true >(abuf, xbuf, wt + 2 * DHID * DHID, b21);
  __syncthreads();
  gemm_tile<false>(abuf, xbuf, wt + 3 * DHID * DHID, b22);
  __syncthreads();

  // ---- coalesced writeback ----
#pragma unroll 4
  for (int t = 0; t < MBLK; ++t)
    out[(size_t)(base + t) * DHID + tid] = xbuf[t * XPITCH + tid];
}

// Convert the four fp32 weight matrices to bf16, transposed to [n][k].
__global__ void convert_weights(const float* __restrict__ W11,
                                const float* __restrict__ W12,
                                const float* __restrict__ W21,
                                const float* __restrict__ W22,
                                bf16* __restrict__ wt) {
  const int id  = blockIdx.x * blockDim.x + threadIdx.x;  // 0 .. 4*65536-1
  const int mat = id >> 16;
  const int r   = id & 0xFFFF;
  const int n   = r >> 8;
  const int k   = r & 255;
  const float* W = (mat == 0) ? W11 : (mat == 1) ? W12 : (mat == 2) ? W21 : W22;
  wt[(size_t)mat * DHID * DHID + n * DHID + k] = (bf16)W[k * DHID + n];
}

extern "C" void kernel_launch(void* const* d_in, const int* in_sizes, int n_in,
                              void* d_out, int out_size, void* d_ws, size_t ws_size,
                              hipStream_t stream) {
  const float* meas        = (const float*)d_in[0];
  const float* evnt        = (const float*)d_in[1];
  const float* leak        = (const float*)d_in[2];
  const float* evlk        = (const float*)d_in[3];
  const int*   sid         = (const int*)  d_in[4];
  const int*   cid         = (const int*)  d_in[5];
  const float* w_meas      = (const float*)d_in[6];
  const float* b_meas      = (const float*)d_in[7];
  const float* w_event     = (const float*)d_in[8];
  const float* b_event     = (const float*)d_in[9];
  const float* w_leak      = (const float*)d_in[10];
  const float* b_leak      = (const float*)d_in[11];
  const float* w_el        = (const float*)d_in[12];
  const float* b_el        = (const float*)d_in[13];
  const float* stab_table  = (const float*)d_in[14];
  const float* cycle_table = (const float*)d_in[15];
  const float* g1  = (const float*)d_in[16];
  const float* be1 = (const float*)d_in[17];
  const float* W11 = (const float*)d_in[18];
  const float* b11 = (const float*)d_in[19];
  const float* W12 = (const float*)d_in[20];
  const float* b12 = (const float*)d_in[21];
  const float* g2  = (const float*)d_in[22];
  const float* be2 = (const float*)d_in[23];
  const float* W21 = (const float*)d_in[24];
  const float* b21 = (const float*)d_in[25];
  const float* W22 = (const float*)d_in[26];
  const float* b22 = (const float*)d_in[27];

  bf16* wt = (bf16*)d_ws;   // needs 4*256*256*2 = 512 KB of scratch

  // 1) weights fp32 -> bf16 transposed (L2-resident afterwards)
  convert_weights<<<(4 * DHID * DHID) / 256, 256, 0, stream>>>(W11, W12, W21, W22, wt);

  // 2) fused embedding + 2 residual MLP blocks
  const int ntok = in_sizes[0];           // B*S
  stab_embed_resmlp<<<ntok / MBLK, 256, 0, stream>>>(
      meas, evnt, leak, evlk, sid, cid,
      w_meas, b_meas, w_event, b_event, w_leak, b_leak, w_el, b_el,
      stab_table, cycle_table,
      g1, be1, b11, b12, g2, be2, b21, b22,
      wt, (float*)d_out);
}
